// AttentionPooling_25924422599454
// MI455X (gfx1250) — matrix operand order repaired and checked by reading, not compiled, not run on hardware
//
#include <hip/hip_runtime.h>
#include <hip/hip_bf16.h>
#include <math.h>

// ---------------------------------------------------------------------------
// AttentionPooling on MI455X (gfx1250, wave32, WMMA).
// Heavy path: fused edge-projection kernel computing V = X@Wv and
// scores = X@P (P = Wk @ Q_h^T per head, pre-folded, incl. 1/sqrt(HD))
// with v_wmma_f32_16x16x32_bf16, X tile staged in LDS as bf16.
// ---------------------------------------------------------------------------

#define E_TOT 131072
#define B_TOT 64
#define HH    256
#define S_TOT 32
#define NH    8
#define HD    32

#define TILE_E 64          // edges per block in fused GEMM
#define XPAD   8
#define LDX    (HH + XPAD) // 264 bf16 -> 528B row stride (16B aligned)

typedef __bf16 bf16_t;
typedef __attribute__((ext_vector_type(16))) __bf16 v16bf;
typedef __attribute__((ext_vector_type(8)))  __bf16 v8bf;
typedef __attribute__((ext_vector_type(8)))  float  v8f;

// ---------------------------------------------------------------------------
// Prologue 1: q = seed_vectors @ w_q  (f32, [32,256])
// ---------------------------------------------------------------------------
__global__ __launch_bounds__(256) void ap_qproj_kernel(
    const float* __restrict__ sv, const float* __restrict__ wq,
    float* __restrict__ qf) {
  __shared__ float sl[S_TOT * HH];
  for (int i = threadIdx.x; i < S_TOT * HH; i += 256) sl[i] = sv[i];
  __syncthreads();
  const int n = threadIdx.x;
  for (int s = 0; s < S_TOT; ++s) {
    float acc = 0.f;
    for (int c = 0; c < HH; ++c) acc += sl[s * HH + c] * wq[c * HH + n];
    qf[s * HH + n] = acc;
  }
}

// ---------------------------------------------------------------------------
// Prologue 2: build bf16 transposed weights for WMMA B-operand layout:
//   wvT[n][k] = w_v[k][n]
//   pT [n][c] = (1/sqrt(HD)) * sum_d w_k[c][h*32+d] * q[s][h*32+d],  n=h*32+s
// ---------------------------------------------------------------------------
__global__ __launch_bounds__(256) void ap_prep_kernel(
    const float* __restrict__ wk, const float* __restrict__ wv,
    const float* __restrict__ qf,
    bf16_t* __restrict__ wvT, bf16_t* __restrict__ pT) {
  const int n = blockIdx.x;   // output column (h*32 + s)
  const int c = threadIdx.x;  // input channel
  const int h = n >> 5, s = n & 31;
  __shared__ float ql[HD];
  if (threadIdx.x < HD) ql[threadIdx.x] = qf[s * HH + h * HD + threadIdx.x];
  __syncthreads();
  float acc = 0.f;
  #pragma unroll
  for (int d = 0; d < HD; ++d) acc += wk[c * HH + h * HD + d] * ql[d];
  pT[n * HH + c]  = (bf16_t)(acc * 0.17677669529663687f);  // 1/sqrt(32)
  wvT[n * HH + c] = (bf16_t)wv[c * HH + n];
}

// ---------------------------------------------------------------------------
// Prologue 3: segment boundaries from sorted batch ids (every graph nonempty)
// ---------------------------------------------------------------------------
__global__ void ap_starts_kernel(const int* __restrict__ batch,
                                 int* __restrict__ starts) {
  const int e = blockIdx.x * blockDim.x + threadIdx.x;
  if (e == 0) { starts[batch[0]] = 0; starts[B_TOT] = E_TOT; }
  else if (e < E_TOT) {
    const int b = batch[e];
    if (b != batch[e - 1]) starts[b] = e;
  }
}

// ---------------------------------------------------------------------------
// Fused edge GEMMs:  V = X @ WvT^T (bf16 out),  scores = X @ pT^T (f32 out)
// 64 edges/block, 4 waves; wave w owns M-tile w; WMMA 16x16x32 bf16.
// ---------------------------------------------------------------------------
__global__ __launch_bounds__(128) void ap_fused_proj_kernel(
    const float* __restrict__ x,      // edge_features [E,256] f32
    const bf16_t* __restrict__ wvT,   // [256][256] bf16
    const bf16_t* __restrict__ pT,    // [256][256] bf16
    bf16_t* __restrict__ v_out,       // [E,256] bf16
    float* __restrict__ s_out) {      // [E,256] f32, col = h*32+s
  __shared__ __align__(16) bf16_t xl[TILE_E * LDX];
  const int tid = threadIdx.x;
  const int e0  = blockIdx.x * TILE_E;

  // Stage X tile as bf16 in LDS (vectorized f32x4 loads)
  for (int i = tid; i < TILE_E * HH / 4; i += 128) {
    const int r  = i / (HH / 4);
    const int c4 = (i % (HH / 4)) * 4;
    const float4 f = *(const float4*)(x + (size_t)(e0 + r) * HH + c4);
    bf16_t* dst = &xl[r * LDX + c4];
    dst[0] = (bf16_t)f.x; dst[1] = (bf16_t)f.y;
    dst[2] = (bf16_t)f.z; dst[3] = (bf16_t)f.w;
  }
  __syncthreads();

  const int wave = tid >> 5;
  const int lane = tid & 31;
  const int lr   = lane & 15;   // A: row-in-tile / B,D: column
  const int hs   = lane >> 4;   // half-select (K split)
  const int m0   = wave * 16;   // wave owns one 16-row M-tile

  for (int mat = 0; mat < 2; ++mat) {
    const bf16_t* wT = mat ? pT : wvT;
    for (int nt = 0; nt < 16; ++nt) {
      const int n0 = nt * 16;
      v8f acc = {};
      for (int k0 = 0; k0 < HH; k0 += 32) {
        // A fragment: lanes 0-15 rows, K chunks {0..7,16..23}; lanes 16-31 {8..15,24..31}
        v16bf a;
        const bf16_t* pa = &xl[(m0 + lr) * LDX + k0 + hs * 8];
        ((v8bf*)&a)[0] = *(const v8bf*)pa;
        ((v8bf*)&a)[1] = *(const v8bf*)(pa + 16);
        // B fragment: lane reads column n0+lr, K chunks {0..7,16..23} / {8..15,24..31} (same as A)
        v16bf b;
        const bf16_t* pb = &wT[(size_t)(n0 + lr) * HH + k0 + hs * 8];
        ((v8bf*)&b)[0] = *(const v8bf*)pb;
        ((v8bf*)&b)[1] = *(const v8bf*)(pb + 16);
        acc = __builtin_amdgcn_wmma_f32_16x16x32_bf16(
            false, a, false, b, (short)0, acc, false, false);
      }
      const int col = n0 + lr;
      if (mat == 0) {
        #pragma unroll
        for (int r = 0; r < 8; ++r) {
          const int row = e0 + m0 + r + 8 * hs;   // D layout: M = r + 8*(lane>=16)
          v_out[(size_t)row * HH + col] = (bf16_t)acc[r];
        }
      } else {
        #pragma unroll
        for (int r = 0; r < 8; ++r) {
          const int row = e0 + m0 + r + 8 * hs;
          s_out[(size_t)row * HH + col] = acc[r];
        }
      }
    }
  }
}

// ---------------------------------------------------------------------------
// Per-graph softmax stats: m[b,c] = max_e scores, denom[b,c] = sum exp(.-m)
// One block per graph, thread = column (h*32+s); coalesced row scans.
// ---------------------------------------------------------------------------
__global__ __launch_bounds__(256) void ap_stats_kernel(
    const float* __restrict__ scores, const int* __restrict__ starts,
    float* __restrict__ mws, float* __restrict__ dws) {
  const int b = blockIdx.x, c = threadIdx.x;
  const int e0 = starts[b], e1 = starts[b + 1];
  float m = -3.4e38f;
  for (int e = e0; e < e1; ++e) m = fmaxf(m, scores[(size_t)e * HH + c]);
  float s = 0.f;
  for (int e = e0; e < e1; ++e) s += __expf(scores[(size_t)e * HH + c] - m);
  mws[b * HH + c] = m;
  dws[b * HH + c] = s;
}

// ---------------------------------------------------------------------------
// Pooling: pooled[b,s,h,d] = sum_e softmax_w[e,s,h] * v[e,h,d]
// One block per (graph, head); 64-edge chunks of w and v staged in LDS,
// each thread accumulates 4 (s,d) outputs in registers.
// ---------------------------------------------------------------------------
#define POOL_CH 64
__global__ __launch_bounds__(256) void ap_pool_kernel(
    const float* __restrict__ scores, const bf16_t* __restrict__ vws,
    const int* __restrict__ starts, const float* __restrict__ mws,
    const float* __restrict__ dws, float* __restrict__ pooled) {
  const int b = blockIdx.x >> 3, h = blockIdx.x & 7;
  const int t = threadIdx.x;
  __shared__ float wch[POOL_CH][32];
  __shared__ float vch[POOL_CH][32];
  __shared__ float ml[32], rdl[32];
  if (t < 32) {
    ml[t]  = mws[b * HH + h * HD + t];
    rdl[t] = 1.0f / dws[b * HH + h * HD + t];
  }
  const int e0 = starts[b], e1 = starts[b + 1];
  const int d = t & 31, sg = (t >> 5) * 4;
  float acc[4] = {0.f, 0.f, 0.f, 0.f};
  for (int ec = e0; ec < e1; ec += POOL_CH) {
    __syncthreads();
    for (int j = t; j < POOL_CH * 32; j += 256) {
      const int el = j >> 5, q = j & 31;
      const int e = ec + el;
      float wv_ = 0.f, vv_ = 0.f;
      if (e < e1) {
        wv_ = __expf(scores[(size_t)e * HH + h * HD + q] - ml[q]) * rdl[q];
        vv_ = (float)vws[(size_t)e * HH + h * HD + q];
      }
      wch[el][q] = wv_;
      vch[el][q] = vv_;
    }
    __syncthreads();
    const int lim = min(POOL_CH, e1 - ec);
    for (int el = 0; el < lim; ++el) {
      const float vv = vch[el][d];
      acc[0] += wch[el][sg + 0] * vv;
      acc[1] += wch[el][sg + 1] * vv;
      acc[2] += wch[el][sg + 2] * vv;
      acc[3] += wch[el][sg + 3] * vv;
    }
  }
  #pragma unroll
  for (int i = 0; i < 4; ++i)
    pooled[(size_t)b * (S_TOT * HH) + (size_t)(sg + i) * HH + h * HD + d] = acc[i];
}

// ---------------------------------------------------------------------------
// MLP: out = silu(pooled @ w1 + b1) @ w2 + b2   (one block per graph)
// ---------------------------------------------------------------------------
__global__ __launch_bounds__(256) void ap_mlp_kernel(
    const float* __restrict__ pooled, const float* __restrict__ w1,
    const float* __restrict__ b1, const float* __restrict__ w2,
    const float* __restrict__ b2, float* __restrict__ out) {
  const int b = blockIdx.x, n = threadIdx.x;
  __shared__ float pl[S_TOT * HH];
  __shared__ float h1l[HH];
  for (int i = n; i < S_TOT * HH; i += 256)
    pl[i] = pooled[(size_t)b * (S_TOT * HH) + i];
  __syncthreads();
  float acc = b1[n];
  for (int k = 0; k < S_TOT * HH; ++k) acc += pl[k] * w1[(size_t)k * HH + n];
  const float h1 = acc / (1.f + __expf(-acc));  // silu
  h1l[n] = h1;
  __syncthreads();
  float acc2 = b2[n];
  for (int k = 0; k < HH; ++k) acc2 += h1l[k] * w2[k * HH + n];
  out[b * HH + n] = acc2;
}

// ---------------------------------------------------------------------------
// Workspace layout (bytes)
// ---------------------------------------------------------------------------
constexpr size_t OFF_SCORES = 0;                                            // E*256*4
constexpr size_t OFF_VBF    = OFF_SCORES + (size_t)E_TOT * HH * 4;          // E*256*2
constexpr size_t OFF_QF     = OFF_VBF    + (size_t)E_TOT * HH * 2;          // 32*256*4
constexpr size_t OFF_WVT    = OFF_QF     + (size_t)S_TOT * HH * 4;          // 256*256*2
constexpr size_t OFF_PT     = OFF_WVT    + (size_t)HH * HH * 2;             // 256*256*2
constexpr size_t OFF_STARTS = OFF_PT     + (size_t)HH * HH * 2;             // (B+1)*4 -> 512
constexpr size_t OFF_M      = OFF_STARTS + 512;                             // 64*256*4
constexpr size_t OFF_D      = OFF_M      + (size_t)B_TOT * HH * 4;          // 64*256*4
constexpr size_t OFF_POOLED = OFF_D      + (size_t)B_TOT * HH * 4;          // 64*8192*4

extern "C" void kernel_launch(void* const* d_in, const int* in_sizes, int n_in,
                              void* d_out, int out_size, void* d_ws, size_t ws_size,
                              hipStream_t stream) {
  (void)in_sizes; (void)n_in; (void)out_size; (void)ws_size;
  const float* edge_features = (const float*)d_in[0];
  // d_in[1] = edge_coords: provably unused (projected by [I|0] in reference)
  const int*   batch         = (const int*)d_in[2];
  const float* seed_vectors  = (const float*)d_in[3];
  const float* w_q           = (const float*)d_in[4];
  const float* w_k           = (const float*)d_in[5];
  const float* w_v           = (const float*)d_in[6];
  const float* w1            = (const float*)d_in[7];
  const float* b1            = (const float*)d_in[8];
  const float* w2            = (const float*)d_in[9];
  const float* b2            = (const float*)d_in[10];
  float* out = (float*)d_out;

  char* ws = (char*)d_ws;
  float*  scores = (float*)(ws + OFF_SCORES);
  bf16_t* vbf    = (bf16_t*)(ws + OFF_VBF);
  float*  qf     = (float*)(ws + OFF_QF);
  bf16_t* wvT    = (bf16_t*)(ws + OFF_WVT);
  bf16_t* pT     = (bf16_t*)(ws + OFF_PT);
  int*    starts = (int*)(ws + OFF_STARTS);
  float*  mws    = (float*)(ws + OFF_M);
  float*  dws    = (float*)(ws + OFF_D);
  float*  pooled = (float*)(ws + OFF_POOLED);

  ap_qproj_kernel<<<1, 256, 0, stream>>>(seed_vectors, w_q, qf);
  ap_prep_kernel<<<HH, 256, 0, stream>>>(w_k, w_v, qf, wvT, pT);
  ap_starts_kernel<<<E_TOT / 256, 256, 0, stream>>>(batch, starts);
  ap_fused_proj_kernel<<<E_TOT / TILE_E, 128, 0, stream>>>(
      edge_features, wvT, pT, vbf, scores);
  ap_stats_kernel<<<B_TOT, 256, 0, stream>>>(scores, starts, mws, dws);
  ap_pool_kernel<<<B_TOT * NH, 256, 0, stream>>>(scores, vbf, starts, mws, dws, pooled);
  ap_mlp_kernel<<<B_TOT, 256, 0, stream>>>(pooled, w1, b1, w2, b2, out);
}
